// InstrumentedLOCMBlock_23407571763714
// MI455X (gfx1250) — matrix operand order repaired and checked by reading, not compiled, not run on hardware
//
#include <hip/hip_runtime.h>

typedef __attribute__((ext_vector_type(16))) _Float16 v16h;
typedef __attribute__((ext_vector_type(8)))  float    v8f;

#define BATCH  4096
#define RDIM   1024
#define UDIM   512
#define HIDDEN 1024
#define FEATS  2560           // 2*RDIM + UDIM
#define GAMMA_MAX 2.0f

#define BM 128
#define BN 128
#define BK 32
#define LDSS 40               // padded LDS row stride (halfs): 80B rows, 16B-aligned

#define EP_NONE     0
#define EP_TANH_F16 1
#define EP_SIG2     2

// --- CDNA5 async global->LDS path (guarded; falls back to VGPR staging) ----
#if defined(__has_builtin)
#  if __has_builtin(__builtin_amdgcn_global_load_async_to_lds_b128)
#    define ASYNC_LDS 1
#  endif
#  if __has_builtin(__builtin_amdgcn_s_wait_asynccnt)
#    define HAVE_WAIT_ASYNC 1
#  endif
#endif

// b128 payload type matching the builtin's parameter: int __vector(4)*
typedef int b128_t __attribute__((vector_size(4 * sizeof(int))));
#define ASL __attribute__((address_space(3)))

__device__ __forceinline__ void wait_async_zero()
{
#if defined(HAVE_WAIT_ASYNC)
    __builtin_amdgcn_s_wait_asynccnt(0);
#elif defined(ASYNC_LDS)
    asm volatile("s_wait_asynccnt 0x0" ::: "memory");
#endif
}

// After issuing 8 prefetch loads for the NEXT tile, <=8 outstanding means the
// CURRENT tile's 8 loads (issued one iteration earlier) have completed
// (async loads return "done" in order, ISA 08 §4.1).
__device__ __forceinline__ void wait_async_le8()
{
#if defined(HAVE_WAIT_ASYNC)
    __builtin_amdgcn_s_wait_asynccnt(8);
#elif defined(ASYNC_LDS)
    asm volatile("s_wait_asynccnt 0x8" ::: "memory");
#endif
}

// ---------------------------------------------------------------------------
// Tiled f16 WMMA GEMM: out = A[M,K] @ W[N,K]^T (+bias), M = gridDim.y*BM.
// 128 threads = 4 waves; wave (wm,wn) computes a 64x64 C tile = 4x4 WMMA tiles
// -> 16 v_wmma_f32_16x16x32_f16 per K-step per wave.
// Double-buffered LDS with one-K-step async prefetch.
// ---------------------------------------------------------------------------
template<int EP>
__global__ __launch_bounds__(128)
void gemm_f16_wmma(const _Float16* __restrict__ A, int lda,
                   const _Float16* __restrict__ W, int ldw,
                   const float* __restrict__ bias,
                   float* __restrict__ outF, _Float16* __restrict__ outH,
                   int N, int K)
{
    __shared__ __align__(16) _Float16 sA[2][BM * LDSS];
    __shared__ __align__(16) _Float16 sB[2][BN * LDSS];

    const int tid    = threadIdx.x;
    const int lane   = tid & 31;          // wave32
    const int wid    = tid >> 5;          // 4 waves
    const int lh     = lane & 15;
    const int hi     = lane >> 4;         // 0: lanes 0-15, 1: lanes 16-31
    const int bm     = blockIdx.y;
    const int bn     = blockIdx.x;
    const int wave_m = wid & 1;           // 2 wave rows (64 rows each)
    const int wave_n = wid >> 1;          // 2 wave cols (64 cols each)

    v8f acc[4][4];
    #pragma unroll
    for (int mf = 0; mf < 4; ++mf)
        #pragma unroll
        for (int nf = 0; nf < 4; ++nf) { v8f z = {}; acc[mf][nf] = z; }

    const _Float16* Ag = A + (size_t)(bm * BM) * lda;
    const _Float16* Wg = W + (size_t)(bn * BN) * ldw;

    // Issue one tile-pair (A+W) of async copies into buffer `buf`:
    // 128x32 halfs per tile = 512 uint4; 128 threads x 4 each; 8 instr/wave.
    auto issue_tile = [&](int buf, int kk) {
        #pragma unroll
        for (int i = 0; i < 4; ++i) {
            const int idx = tid + i * 128;
            const int row = idx >> 2;          // 4 uint4 per 32-half row
            const int col = (idx & 3) << 3;    // 8 halfs per uint4
            const _Float16* ga = Ag + (size_t)row * lda + kk + col;
            const _Float16* gw = Wg + (size_t)row * ldw + kk + col;
            _Float16* la = &sA[buf][row * LDSS + col];
            _Float16* lb = &sB[buf][row * LDSS + col];
#if defined(ASYNC_LDS)
            __builtin_amdgcn_global_load_async_to_lds_b128(
                (b128_t*)ga, (ASL b128_t*)la, 0, 0);
            __builtin_amdgcn_global_load_async_to_lds_b128(
                (b128_t*)gw, (ASL b128_t*)lb, 0, 0);
#else
            *(uint4*)la = *(const uint4*)ga;
            *(uint4*)lb = *(const uint4*)gw;
#endif
        }
    };

    issue_tile(0, 0);                      // prologue prefetch of tile 0

    int buf = 0;
    for (int k0 = 0; k0 < K; k0 += BK, buf ^= 1) {
        if (k0 + BK < K) {
            // Prefetch next tile into the other buffer (free since the
            // trailing barrier of iteration k0-BK), then wait only for the
            // CURRENT tile's loads.
            issue_tile(buf ^ 1, k0 + BK);
            wait_async_le8();
        } else {
            wait_async_zero();
        }
        __syncthreads();                   // current tile visible to all waves

        // A fragment (ISA 16-bit 16x32 layout): row = lh; K = hi*8 + {0..7,16..23}
        v16h a[4], b[4];
        #pragma unroll
        for (int mf = 0; mf < 4; ++mf) {
            const int base = (wave_m * 64 + mf * 16 + lh) * LDSS + hi * 8;
            *((float4*)&a[mf])     = *(const float4*)&sA[buf][base];
            *((float4*)&a[mf] + 1) = *(const float4*)&sA[buf][base + 16];
        }
        // B fragment: col = lh; K = hi*8 + {0..7,16..23}
        #pragma unroll
        for (int nf = 0; nf < 4; ++nf) {
            const int base = (wave_n * 64 + nf * 16 + lh) * LDSS + hi * 8;
            *((float4*)&b[nf])     = *(const float4*)&sB[buf][base];
            *((float4*)&b[nf] + 1) = *(const float4*)&sB[buf][base + 16];
        }

        #pragma unroll
        for (int nf = 0; nf < 4; ++nf)
            #pragma unroll
            for (int mf = 0; mf < 4; ++mf)
                acc[mf][nf] = __builtin_amdgcn_wmma_f32_16x16x32_f16(
                    false, a[mf], false, b[nf], (short)0, acc[mf][nf], false, false);

        __syncthreads();                   // all waves done reading `buf`
    }

    // Epilogue. C/D layout: VGPR j -> row = j + hi*8 ; col = lh.
    int   coln[4];
    float bs[4];
    #pragma unroll
    for (int nf = 0; nf < 4; ++nf) {
        coln[nf] = bn * BN + wave_n * 64 + nf * 16 + lh;
        bs[nf]   = bias ? bias[coln[nf]] : 0.0f;
    }
    const int rowb = bm * BM + wave_m * 64;
    #pragma unroll
    for (int mf = 0; mf < 4; ++mf)
        #pragma unroll
        for (int nf = 0; nf < 4; ++nf)
            #pragma unroll
            for (int j = 0; j < 8; ++j) {
                const int    row = rowb + mf * 16 + hi * 8 + j;
                const size_t off = (size_t)row * (size_t)N + coln[nf];
                const float  v   = acc[mf][nf][j] + bs[nf];
                if (EP == EP_NONE) {
                    outF[off] = v;
                } else if (EP == EP_TANH_F16) {
                    outH[off] = (_Float16)tanhf(v);
                } else { // EP_SIG2: gamma = GAMMA_MAX * sigmoid(v)
                    outF[off] = GAMMA_MAX / (1.0f + __expf(-v));
                }
            }
}

// ---------------------------------------------------------------------------
// Elementwise helpers
// ---------------------------------------------------------------------------
__global__ __launch_bounds__(256)
void f32_to_f16(const float* __restrict__ in, _Float16* __restrict__ out, int n)
{
    const int i = blockIdx.x * 256 + threadIdx.x;
    if (i < n) out[i] = (_Float16)in[i];
}

__global__ __launch_bounds__(256)
void pack_feats(const float* __restrict__ y, const float* __restrict__ z,
                const float* __restrict__ u, _Float16* __restrict__ feats)
{
    const int i = blockIdx.x * 256 + threadIdx.x;   // over BATCH*FEATS
    const int b = i / FEATS;
    const int c = i - b * FEATS;
    float v;
    if (c < RDIM)            v = y[(size_t)b * RDIM + c];
    else if (c < 2 * RDIM)   v = z[(size_t)b * RDIM + (c - RDIM)];
    else                     v = u[(size_t)b * UDIM + (c - 2 * RDIM)];
    feats[i] = (_Float16)v;
}

__global__ __launch_bounds__(256)
void osc_step(const float* __restrict__ y, const float* __restrict__ z,
              const float* __restrict__ Bdu, const float* __restrict__ gamma,
              const float* __restrict__ omega_sq, const float* __restrict__ alpha_lambda,
              const float* __restrict__ dtp,
              float* __restrict__ z_next, float* __restrict__ y_next,
              _Float16* __restrict__ ynextH)
{
    const int i  = blockIdx.x * 256 + threadIdx.x;  // over BATCH*RDIM
    const int r  = i & (RDIM - 1);
    const float dt  = *dtp;
    const float omc = omega_sq[r] - alpha_lambda[r];
    const float zi  = z[i];
    const float yi  = y[i];
    const float zn  = zi + dt * (Bdu[i] - omc * yi - gamma[i] * zi);
    const float yn  = yi + dt * zn;
    z_next[i] = zn;
    y_next[i] = yn;
    ynextH[i] = (_Float16)yn;
}

__global__ __launch_bounds__(256)
void glu_combine(const float* __restrict__ s1, const float* __restrict__ s2,
                 float* __restrict__ yout)
{
    const int i = blockIdx.x * 256 + threadIdx.x;   // over BATCH*RDIM
    yout[i] = s2[i] / (1.0f + __expf(-s1[i]));
}

// ---------------------------------------------------------------------------
// Launch: inputs in setup_inputs() order; outputs (y_out, z_next, gamma, y_next)
// concatenated flat in d_out. Workspace layout (~64 MB, with aliasing).
// ---------------------------------------------------------------------------
extern "C" void kernel_launch(void* const* d_in, const int* in_sizes, int n_in,
                              void* d_out, int out_size, void* d_ws, size_t ws_size,
                              hipStream_t stream)
{
    const float* y_tilde   = (const float*)d_in[0];
    const float* z_tilde   = (const float*)d_in[1];
    const float* u         = (const float*)d_in[2];
    const float* dtp       = (const float*)d_in[3];
    const float* omega_sq  = (const float*)d_in[4];
    const float* alpha_lam = (const float*)d_in[5];
    const float* B_tilde   = (const float*)d_in[6];
    const float* w1        = (const float*)d_in[7];
    const float* b1        = (const float*)d_in[8];
    const float* w2        = (const float*)d_in[9];
    const float* b2        = (const float*)d_in[10];
    const float* gw1       = (const float*)d_in[11];
    const float* gw2       = (const float*)d_in[12];
    const float* gb1       = (const float*)d_in[13];
    const float* gb2       = (const float*)d_in[14];

    float* out = (float*)d_out;
    const size_t BR = (size_t)BATCH * RDIM;
    float* out_yout  = out;
    float* out_znext = out + BR;
    float* out_gamma = out + 2 * BR;
    float* out_ynext = out + 3 * BR;

    char* ws = (char*)d_ws;
    _Float16* featsH = (_Float16*)(ws + 0);            // 4096*2560*2 = 20,971,520
    _Float16* hH     = (_Float16*)(ws + 20971520);     // 4096*1024*2 =  8,388,608
    _Float16* ynH    = (_Float16*)(ws + 29360128);     //                8,388,608
    float*    Bdu    = (float*)   (ws + 37748736);     // 4096*1024*4 = 16,777,216
    float*    s1     = Bdu;                            // alias: Bdu dead after osc_step
    float*    s2     = (float*)   (ws + 0);            // alias: feats dead after GEMM1
    _Float16* W1h    = (_Float16*)(ws + 54525952);     // 1024*2560*2 =  5,242,880
    _Float16* Bth    = (_Float16*)(ws + 59768832);     // 1024* 512*2 =  1,048,576
    _Float16* W2h    = (_Float16*)(ws + 60817408);     // 1024*1024*2 =  2,097,152
    _Float16* G1h    = (_Float16*)(ws + 62914560);     //                2,097,152
    _Float16* G2h    = (_Float16*)(ws + 65011712);     //                2,097,152

    // 1) Weight conversions to f16
    f32_to_f16<<<(RDIM * UDIM    + 255) / 256, 256, 0, stream>>>(B_tilde, Bth, RDIM * UDIM);
    f32_to_f16<<<(HIDDEN * FEATS + 255) / 256, 256, 0, stream>>>(w1, W1h, HIDDEN * FEATS);
    f32_to_f16<<<(RDIM * HIDDEN  + 255) / 256, 256, 0, stream>>>(w2, W2h, RDIM * HIDDEN);
    f32_to_f16<<<(RDIM * RDIM    + 255) / 256, 256, 0, stream>>>(gw1, G1h, RDIM * RDIM);
    f32_to_f16<<<(RDIM * RDIM    + 255) / 256, 256, 0, stream>>>(gw2, G2h, RDIM * RDIM);

    // 2) Pack feats = [y | z | u] as f16
    pack_feats<<<(BATCH * FEATS) / 256, 256, 0, stream>>>(y_tilde, z_tilde, u, featsH);

    const dim3 gg(RDIM / BN, BATCH / BM);   // (8, 32), 4 waves/block

    // 3) B_dot_u = u @ B_tilde^T  (A = u-columns inside feats, lda = FEATS)
    gemm_f16_wmma<EP_NONE><<<gg, 128, 0, stream>>>(
        featsH + 2 * RDIM, FEATS, Bth, UDIM, nullptr, Bdu, nullptr, RDIM, UDIM);

    // 4) h = tanh(feats @ w1^T + b1) -> f16
    gemm_f16_wmma<EP_TANH_F16><<<gg, 128, 0, stream>>>(
        featsH, FEATS, W1h, FEATS, b1, nullptr, hH, HIDDEN, FEATS);

    // 5) gamma = 2*sigmoid(h @ w2^T + b2) -> d_out section 2
    gemm_f16_wmma<EP_SIG2><<<gg, 128, 0, stream>>>(
        hH, HIDDEN, W2h, HIDDEN, b2, out_gamma, nullptr, RDIM, HIDDEN);

    // 6) semi-implicit Euler step: z_next, y_next (+f16 copy for GLU)
    osc_step<<<(BATCH * RDIM) / 256, 256, 0, stream>>>(
        y_tilde, z_tilde, Bdu, out_gamma, omega_sq, alpha_lam, dtp,
        out_znext, out_ynext, ynH);

    // 7) GLU: s1 = y_next@gw1^T + gb1 ; s2 = y_next@gw2^T + gb2
    gemm_f16_wmma<EP_NONE><<<gg, 128, 0, stream>>>(
        ynH, RDIM, G1h, RDIM, gb1, s1, nullptr, RDIM, RDIM);
    gemm_f16_wmma<EP_NONE><<<gg, 128, 0, stream>>>(
        ynH, RDIM, G2h, RDIM, gb2, s2, nullptr, RDIM, RDIM);

    // 8) y_out = sigmoid(s1) * s2
    glu_combine<<<(BATCH * RDIM) / 256, 256, 0, stream>>>(s1, s2, out_yout);
}
